// Graph_Prop_32650341384595
// MI455X (gfx1250) — matrix operand order repaired and checked by reading, not compiled, not run on hardware
//
#include <hip/hip_runtime.h>
#include <hip/hip_bf16.h>

#define H_IN 228
#define W_IN 304
#define H_G  76
#define W_G  102
#define NN   (H_G * W_G)   // 7752 nodes
#define NP   7760          // padded to multiple of 16
#define NT   (NP / 16)     // 485 tiles
#define KNN_K 16
#define KP   96            // padded contraction dim (81->96, 96->96)
#define NB   2             // batch

typedef __attribute__((ext_vector_type(16))) __bf16 v16bf;
typedef __attribute__((ext_vector_type(8)))  float  v8f;

union UB { v16bf v; unsigned u[8]; };

// ---------------------------------------------------------------------------
// Build node features: loc (x*d*3, y*d*3, d, |.|^2) and 81 guidance channels
// stored node-major bf16 padded to KP=96.
// ---------------------------------------------------------------------------
__global__ void k_build(const float* __restrict__ guidance,
                        const float* __restrict__ ini,
                        const float* __restrict__ sparse,
                        float4* __restrict__ loc4,
                        __bf16* __restrict__ X0b) {
    int t = blockIdx.x * blockDim.x + threadIdx.x;
    if (t >= NB * NN) return;
    int b = t / NN, n = t % NN;
    int i = n / W_G, j = n % W_G;
    int r = 1 + 3 * i, cidx = 3 * j;
    size_t off = (size_t)b * H_IN * W_IN + (size_t)r * W_IN + cidx;
    float s = sparse[off];
    float mask = (s > 0.f) ? 1.f : ((s < 0.f) ? -1.f : 0.f);
    float d = (1.f - mask) * ini[off] + mask * s;
    const float fx = 582.6244816773795f * 0.5f, fy = 582.6910327098864f * 0.5f;
    const float cx = 313.0447587080473f * 0.5f, cy = 238.44389626620386f * 0.5f;
    float x3 = ((float)j - cx) / fx, y3 = ((float)i - cy) / fy;
    float lx = x3 * d * 3.f, ly = y3 * d * 3.f, lz = d;
    loc4[t] = make_float4(lx, ly, lz, lx * lx + ly * ly + lz * lz);

    __bf16* dst = X0b + ((size_t)b * NP + n) * KP;
    #pragma unroll 1
    for (int c = 0; c < 81; ++c) {
        int rr = (c / 9) - 3 + 3 * i;
        int ss = (c % 9) - 4 + 3 * j;
        float v = 0.f;
        if (rr >= 0 && rr < H_IN && ss >= 0 && ss < W_IN)
            v = guidance[(((size_t)b * 81 + c) * H_IN + rr) * W_IN + ss];
        dst[c] = (__bf16)v;
    }
    for (int c = 81; c < KP; ++c) dst[c] = (__bf16)0.f;
}

// Zero the padded tail nodes of a node-major bf16 feature matrix.
__global__ void k_ztail(__bf16* __restrict__ X) {
    int t = blockIdx.x * blockDim.x + threadIdx.x;
    int per = (NP - NN) * KP;
    if (t >= NB * per) return;
    int b = t / per, rem = t % per;
    int pn = NN + rem / KP, c = rem % KP;
    X[((size_t)b * NP + pn) * KP + c] = (__bf16)0.f;
}

// Split weight W[M x 2F] into Wp = (L - R), Wq = R as bf16, padded to Mp x KP;
// also emit zero-padded bias.
__global__ void k_prepw(const float* __restrict__ W, const float* __restrict__ bias,
                        int Mact, int F,
                        __bf16* __restrict__ Wp, __bf16* __restrict__ Wq,
                        float* __restrict__ bp, int Mp) {
    int t = blockIdx.x * blockDim.x + threadIdx.x;
    if (t >= Mp * KP) return;
    int m = t / KP, k = t % KP;
    float l = 0.f, r = 0.f;
    if (m < Mact && k < F) {
        l = W[(size_t)m * 2 * F + k];
        r = W[(size_t)m * 2 * F + F + k];
    }
    Wp[t] = (__bf16)(l - r);
    Wq[t] = (__bf16)r;
    if (k == 0) bp[m] = (m < Mact) ? bias[m] : 0.f;
}

// ---------------------------------------------------------------------------
// KNN over 3-D loc features (scalar; d=3 is too small for WMMA).
// ---------------------------------------------------------------------------
__global__ void k_knn3(const float4* __restrict__ loc4, int* __restrict__ idx) {
    int t = blockIdx.x * blockDim.x + threadIdx.x;
    if (t >= NB * NN) return;
    int b = t / NN, n = t % NN;
    const float4* L = loc4 + (size_t)b * NN;
    float4 me = L[n];
    float bd[KNN_K]; int bi[KNN_K];
    #pragma unroll
    for (int k = 0; k < KNN_K; ++k) { bd[k] = 3.0e38f; bi[k] = 0; }
    for (int m = 0; m < NN; ++m) {
        float4 o = L[m];
        float dist = me.w + o.w - 2.f * (me.x * o.x + me.y * o.y + me.z * o.z);
        if (dist < bd[KNN_K - 1]) {
            int j = KNN_K - 1;
            while (j > 0 && bd[j - 1] > dist) { bd[j] = bd[j - 1]; bi[j] = bi[j - 1]; --j; }
            bd[j] = dist; bi[j] = m;
        }
    }
    int* o = idx + (size_t)t * KNN_K;
    #pragma unroll
    for (int k = 0; k < KNN_K; ++k) o[k] = bi[k];
}

// ---------------------------------------------------------------------------
// A-tile loader: rows m = base + (lane&15), 16-bit A 16x32 lane map.
// ---------------------------------------------------------------------------
__device__ __forceinline__ v16bf load_a(const __bf16* __restrict__ base_row,
                                        int kb, int hi) {
    UB A;
    const __bf16* a = base_row + kb + hi * 8;
    #pragma unroll
    for (int v = 0; v < 8; ++v) {
        int k = ((v & 3) * 2) + ((v & 4) ? 16 : 0);
        A.u[v] = *(const unsigned*)(a + k);
    }
    return A.v;
}
// B-tile loader: column n = lane&15 at node base, 16-bit B 32x16 lane map.
__device__ __forceinline__ v16bf load_b(const __bf16* __restrict__ Xb,
                                        int nbase, int kb, int ml, int hi) {
    UB B;
    const __bf16* bb = Xb + (size_t)(nbase + ml) * KP + kb + hi * 8;
    #pragma unroll
    for (int v = 0; v < 8; ++v)
        B.u[v] = *(const unsigned*)(bb + ((v & 3) * 2) + ((v & 4) ? 16 : 0));
    return B.v;
}

#define WMMA_BF16(A, B, C) \
    __builtin_amdgcn_wmma_f32_16x16x32_bf16(false, (A), false, (B), (short)0, (C), false, false)

// ---------------------------------------------------------------------------
// Fused 4-output bf16 WMMA GEMM: {P,Q,Pa,Qa}[node][m] = W? . X[node] (+bias).
// One wave per 16x16 tile; one B load feeds 4 WMMAs.
// ---------------------------------------------------------------------------
__global__ void __launch_bounds__(32)
k_gemm4(const __bf16* __restrict__ Wpe, const __bf16* __restrict__ Wqe,
        const __bf16* __restrict__ Wpa, const __bf16* __restrict__ Wqa,
        const float* __restrict__ bpe, const float* __restrict__ bpa,
        const __bf16* __restrict__ X,
        float* __restrict__ P, float* __restrict__ Q,
        float* __restrict__ Pa, float* __restrict__ Qa, int Cstride) {
    int lane = threadIdx.x;
    int ntile = blockIdx.x, mt = blockIdx.y, b = blockIdx.z;
    int hi = lane >> 4, ml = lane & 15;
    const __bf16* Xb = X + (size_t)b * NP * KP;
    size_t wrow = (size_t)(mt * 16 + ml) * KP;
    v8f aP = {}, aQ = {}, aPa = {}, aQa = {};
    #pragma unroll
    for (int k3 = 0; k3 < 3; ++k3) {
        int kb = k3 * 32;
        v16bf B = load_b(Xb, ntile * 16, kb, ml, hi);
        aP  = WMMA_BF16(load_a(Wpe + wrow, kb, hi), B, aP);
        aQ  = WMMA_BF16(load_a(Wqe + wrow, kb, hi), B, aQ);
        aPa = WMMA_BF16(load_a(Wpa + wrow, kb, hi), B, aPa);
        aQa = WMMA_BF16(load_a(Wqa + wrow, kb, hi), B, aQa);
    }
    int node = ntile * 16 + ml;
    int mbase = mt * 16 + hi * 8;
    size_t ob = ((size_t)b * NP + node) * Cstride + mbase;
    float b0 = bpe[mbase + 0], b1 = bpe[mbase + 1], b2 = bpe[mbase + 2], b3 = bpe[mbase + 3];
    float b4 = bpe[mbase + 4], b5 = bpe[mbase + 5], b6 = bpe[mbase + 6], b7 = bpe[mbase + 7];
    *(float4*)(P + ob)     = make_float4(aP[0] + b0, aP[1] + b1, aP[2] + b2, aP[3] + b3);
    *(float4*)(P + ob + 4) = make_float4(aP[4] + b4, aP[5] + b5, aP[6] + b6, aP[7] + b7);
    b0 = bpa[mbase + 0]; b1 = bpa[mbase + 1]; b2 = bpa[mbase + 2]; b3 = bpa[mbase + 3];
    b4 = bpa[mbase + 4]; b5 = bpa[mbase + 5]; b6 = bpa[mbase + 6]; b7 = bpa[mbase + 7];
    *(float4*)(Pa + ob)     = make_float4(aPa[0] + b0, aPa[1] + b1, aPa[2] + b2, aPa[3] + b3);
    *(float4*)(Pa + ob + 4) = make_float4(aPa[4] + b4, aPa[5] + b5, aPa[6] + b6, aPa[7] + b7);
    *(float4*)(Q + ob)      = make_float4(aQ[0], aQ[1], aQ[2], aQ[3]);
    *(float4*)(Q + ob + 4)  = make_float4(aQ[4], aQ[5], aQ[6], aQ[7]);
    *(float4*)(Qa + ob)     = make_float4(aQa[0], aQa[1], aQa[2], aQa[3]);
    *(float4*)(Qa + ob + 4) = make_float4(aQa[4], aQa[5], aQa[6], aQa[7]);
}

// ---------------------------------------------------------------------------
// Fused Gram + top-16 KNN over 96-d features. One wave per TWO 16-row tiles:
// A tiles hoisted into registers; one B load feeds 2 WMMAs; all 32 lanes own
// a row and maintain a register top-16 via LDS hand-off.
// ---------------------------------------------------------------------------
__global__ void __launch_bounds__(32)
k_knng(const __bf16* __restrict__ X, const float* __restrict__ sqn,
       int* __restrict__ idx) {
    __shared__ float ldsd[512];   // 32 rows x 16 cols of dot products
    __shared__ float ldsq[16];    // column sq-norms for this tile
    int lane = threadIdx.x;
    int b = blockIdx.y;
    int hi = lane >> 4, ml = lane & 15;
    const __bf16* Xb = X + (size_t)b * NP * KP;
    const float* sq = sqn + (size_t)b * NP;
    int rb0 = (blockIdx.x * 2) * 16;
    int rb1 = rb0 + 16; if (rb1 > NP - 16) rb1 = NP - 16;

    // Hoist both A row-tiles (2 x 3 kb-slices = 48 VGPRs of bf16 pairs)
    v16bf A0[3], A1[3];
    const __bf16* ar0 = Xb + (size_t)(rb0 + ml) * KP;
    const __bf16* ar1 = Xb + (size_t)(rb1 + ml) * KP;
    #pragma unroll
    for (int k3 = 0; k3 < 3; ++k3) {
        A0[k3] = load_a(ar0, k3 * 32, hi);
        A1[k3] = load_a(ar1, k3 * 32, hi);
    }

    int myrow = (hi ? rb1 : rb0) + ml;     // lane -> owned row
    bool live = myrow < NN;
    float srow = live ? sq[myrow] : 0.f;
    float bd[KNN_K]; int bi[KNN_K];
    #pragma unroll
    for (int k = 0; k < KNN_K; ++k) { bd[k] = 3.0e38f; bi[k] = 0; }

    for (int ct = 0; ct < NT; ++ct) {
        if (lane < 16) ldsq[lane] = sq[ct * 16 + lane];
        v8f acc0 = {}, acc1 = {};
        #pragma unroll
        for (int k3 = 0; k3 < 3; ++k3) {
            v16bf B = load_b(Xb, ct * 16, k3 * 32, ml, hi);
            acc0 = WMMA_BF16(A0[k3], B, acc0);
            acc1 = WMMA_BF16(A1[k3], B, acc1);
        }
        #pragma unroll
        for (int r = 0; r < 8; ++r) {
            ldsd[(r + 8 * hi) * 16 + ml]        = acc0[r];
            ldsd[(16 + r + 8 * hi) * 16 + ml]   = acc1[r];
        }
        __syncthreads();
        if (live) {
            #pragma unroll 1
            for (int cc = 0; cc < 16; ++cc) {
                int m = ct * 16 + cc;
                float dist = (m < NN) ? (srow + ldsq[cc] - 2.f * ldsd[lane * 16 + cc])
                                      : 3.0e38f;
                if (dist < bd[KNN_K - 1]) {
                    int j = KNN_K - 1;
                    while (j > 0 && bd[j - 1] > dist) {
                        bd[j] = bd[j - 1]; bi[j] = bi[j - 1]; --j;
                    }
                    bd[j] = dist; bi[j] = m;
                }
            }
        }
        __syncthreads();
    }
    if (live) {
        int* o = idx + ((size_t)b * NN + myrow) * KNN_K;
        #pragma unroll
        for (int k = 0; k < KNN_K; ++k) o[k] = bi[k];
    }
}

// ---------------------------------------------------------------------------
// Edge aggregation: e = P[n]+Q[j], a = softmax_k(Pa[n]+Qa[j]), out = sum e*a.
// One block per node, one thread per channel; shfl-based sqnorm reduction.
// ---------------------------------------------------------------------------
__global__ void k_agg(const float* __restrict__ P,  const float* __restrict__ Pa,
                      const float* __restrict__ Q,  const float* __restrict__ Qa,
                      const int* __restrict__ idx, int C,
                      float* __restrict__ OutF, __bf16* __restrict__ OutB,
                      float* __restrict__ sqn) {
    __shared__ float red[4];
    int n = blockIdx.x, b = blockIdx.y, c = threadIdx.x;
    size_t base = ((size_t)b * NP + n) * C;
    const int* ix = idx + ((size_t)b * NN + n) * KNN_K;
    float pe = P[base + c], pa = Pa[base + c];
    float e[KNN_K], a[KNN_K];
    float amax = -3.0e38f;
    #pragma unroll
    for (int k = 0; k < KNN_K; ++k) {
        size_t qb = ((size_t)b * NP + ix[k]) * C + c;
        e[k] = pe + Q[qb];
        a[k] = pa + Qa[qb];
        amax = fmaxf(amax, a[k]);
    }
    float ssum = 0.f, accv = 0.f;
    #pragma unroll
    for (int k = 0; k < KNN_K; ++k) {
        float w = __expf(a[k] - amax);
        ssum += w; accv += e[k] * w;
    }
    float out = accv / ssum;
    if (OutB) OutB[((size_t)b * NP + n) * KP + c] = (__bf16)out;   // C == KP here
    if (OutF) OutF[((size_t)b * NN + n) * C + c] = out;
    if (sqn) {
        float v = out * out;
        #pragma unroll
        for (int m = 16; m >= 1; m >>= 1) v += __shfl_xor(v, m, 32);
        if ((c & 31) == 0) red[c >> 5] = v;
        __syncthreads();
        if (c == 0) sqn[(size_t)b * NP + n] = red[0] + red[1] + red[2];
    }
}

// Pixel-shuffle 9ch -> (3H x 3W), crop columns [1, 1+W_IN).
__global__ void k_out(const float* __restrict__ H3, float* __restrict__ out) {
    int t = blockIdx.x * blockDim.x + threadIdx.x;
    if (t >= NB * H_IN * W_IN) return;
    int b = t / (H_IN * W_IN);
    int rem = t % (H_IN * W_IN);
    int y = rem / W_IN, x = rem % W_IN;
    int i = y / 3, p = y % 3;
    int tt = x + 1, j = tt / 3, q = tt % 3;
    int ch = p * 3 + q;
    out[t] = H3[((size_t)b * NN + (size_t)i * W_G + j) * 16 + ch];
}

// ---------------------------------------------------------------------------
extern "C" void kernel_launch(void* const* d_in, const int* in_sizes, int n_in,
                              void* d_out, int out_size, void* d_ws, size_t ws_size,
                              hipStream_t stream) {
    const float* guidance = (const float*)d_in[0];
    const float* ini      = (const float*)d_in[1];
    const float* sparse   = (const float*)d_in[2];
    const float* e_w1 = (const float*)d_in[3];  const float* e_b1 = (const float*)d_in[4];
    const float* a_w1 = (const float*)d_in[5];  const float* a_b1 = (const float*)d_in[6];
    const float* e_w2 = (const float*)d_in[7];  const float* e_b2 = (const float*)d_in[8];
    const float* a_w2 = (const float*)d_in[9];  const float* a_b2 = (const float*)d_in[10];
    const float* e_w3 = (const float*)d_in[11]; const float* e_b3 = (const float*)d_in[12];
    const float* a_w3 = (const float*)d_in[13]; const float* a_b3 = (const float*)d_in[14];
    float* out = (float*)d_out;

    char* ws = (char*)d_ws;
    size_t off = 0;
    auto alloc = [&](size_t bytes) -> char* {
        off = (off + 255) & ~(size_t)255;
        char* p = ws + off; off += bytes; return p;
    };
    float4* loc4 = (float4*)alloc((size_t)NB * NN * sizeof(float4));
    __bf16* X0b  = (__bf16*)alloc((size_t)NB * NP * KP * 2);
    __bf16* H1b  = (__bf16*)alloc((size_t)NB * NP * KP * 2);
    __bf16* H2b  = (__bf16*)alloc((size_t)NB * NP * KP * 2);
    float*  sqn1 = (float*)alloc((size_t)NB * NP * 4);
    float*  sqn2 = (float*)alloc((size_t)NB * NP * 4);
    int*    idx  = (int*)alloc((size_t)NB * NN * KNN_K * 4);
    float*  P    = (float*)alloc((size_t)NB * NP * KP * 4);
    float*  Pa   = (float*)alloc((size_t)NB * NP * KP * 4);
    float*  Q    = (float*)alloc((size_t)NB * NP * KP * 4);
    float*  Qa   = (float*)alloc((size_t)NB * NP * KP * 4);
    float*  H3   = (float*)alloc((size_t)NB * NN * 16 * 4);
    __bf16* Wp1e = (__bf16*)alloc(96 * KP * 2); __bf16* Wq1e = (__bf16*)alloc(96 * KP * 2);
    __bf16* Wp1a = (__bf16*)alloc(96 * KP * 2); __bf16* Wq1a = (__bf16*)alloc(96 * KP * 2);
    __bf16* Wp2e = (__bf16*)alloc(96 * KP * 2); __bf16* Wq2e = (__bf16*)alloc(96 * KP * 2);
    __bf16* Wp2a = (__bf16*)alloc(96 * KP * 2); __bf16* Wq2a = (__bf16*)alloc(96 * KP * 2);
    __bf16* Wp3e = (__bf16*)alloc(16 * KP * 2); __bf16* Wq3e = (__bf16*)alloc(16 * KP * 2);
    __bf16* Wp3a = (__bf16*)alloc(16 * KP * 2); __bf16* Wq3a = (__bf16*)alloc(16 * KP * 2);
    float* bp1e = (float*)alloc(96 * 4); float* bp1a = (float*)alloc(96 * 4);
    float* bp2e = (float*)alloc(96 * 4); float* bp2a = (float*)alloc(96 * 4);
    float* bp3e = (float*)alloc(16 * 4); float* bp3a = (float*)alloc(16 * 4);
    (void)ws_size; (void)in_sizes; (void)n_in; (void)out_size;

    // Features + weight prep
    k_build<<<(NB * NN + 255) / 256, 256, 0, stream>>>(guidance, ini, sparse, loc4, X0b);
    int ztot = NB * (NP - NN) * KP;
    k_ztail<<<(ztot + 255) / 256, 256, 0, stream>>>(X0b);
    k_ztail<<<(ztot + 255) / 256, 256, 0, stream>>>(H1b);
    k_ztail<<<(ztot + 255) / 256, 256, 0, stream>>>(H2b);
    k_prepw<<<(96 * KP + 255) / 256, 256, 0, stream>>>(e_w1, e_b1, 96, 81, Wp1e, Wq1e, bp1e, 96);
    k_prepw<<<(96 * KP + 255) / 256, 256, 0, stream>>>(a_w1, a_b1, 96, 81, Wp1a, Wq1a, bp1a, 96);
    k_prepw<<<(96 * KP + 255) / 256, 256, 0, stream>>>(e_w2, e_b2, 96, 96, Wp2e, Wq2e, bp2e, 96);
    k_prepw<<<(96 * KP + 255) / 256, 256, 0, stream>>>(a_w2, a_b2, 96, 96, Wp2a, Wq2a, bp2a, 96);
    k_prepw<<<(16 * KP + 255) / 256, 256, 0, stream>>>(e_w3, e_b3, 9, 96, Wp3e, Wq3e, bp3e, 16);
    k_prepw<<<(16 * KP + 255) / 256, 256, 0, stream>>>(a_w3, a_b3, 9, 96, Wp3a, Wq3a, bp3a, 16);

    // KNN on 3-D loc
    k_knn3<<<(NB * NN + 255) / 256, 256, 0, stream>>>(loc4, idx);

    dim3 g96(NT, 6, NB), g16(NT, 1, NB), gk((NT + 1) / 2, NB);

    // Layer 1: 81ch -> 96ch
    k_gemm4<<<g96, 32, 0, stream>>>(Wp1e, Wq1e, Wp1a, Wq1a, bp1e, bp1a, X0b,
                                    P, Q, Pa, Qa, 96);
    k_agg<<<dim3(NN, NB), 96, 0, stream>>>(P, Pa, Q, Qa, idx, 96, nullptr, H1b, sqn1);

    // KNN on H1, layer 2: 96 -> 96
    k_knng<<<gk, 32, 0, stream>>>(H1b, sqn1, idx);
    k_gemm4<<<g96, 32, 0, stream>>>(Wp2e, Wq2e, Wp2a, Wq2a, bp2e, bp2a, H1b,
                                    P, Q, Pa, Qa, 96);
    k_agg<<<dim3(NN, NB), 96, 0, stream>>>(P, Pa, Q, Qa, idx, 96, nullptr, H2b, sqn2);

    // KNN on H2, layer 3: 96 -> 9 (padded to 16)
    k_knng<<<gk, 32, 0, stream>>>(H2b, sqn2, idx);
    k_gemm4<<<g16, 32, 0, stream>>>(Wp3e, Wq3e, Wp3a, Wq3a, bp3e, bp3a, H2b,
                                    P, Q, Pa, Qa, 16);
    k_agg<<<dim3(NN, NB), 16, 0, stream>>>(P, Pa, Q, Qa, idx, 16, H3, nullptr, nullptr);

    // Pixel-shuffle + crop
    k_out<<<(NB * H_IN * W_IN + 255) / 256, 256, 0, stream>>>(H3, out);
}
